// SparseMHAEncoder_69346541961598
// MI455X (gfx1250) — matrix operand order repaired and checked by reading, not compiled, not run on hardware
//
#include <hip/hip_runtime.h>

typedef unsigned short u16;
typedef __bf16 bf16;
typedef __attribute__((ext_vector_type(8)))  bf16  v8bf;
typedef __attribute__((ext_vector_type(16))) bf16  v16bf;
typedef __attribute__((ext_vector_type(8)))  float v8f;
typedef __attribute__((ext_vector_type(4)))  unsigned int v4u;
typedef __attribute__((ext_vector_type(8)))  int v8i_t;
typedef __attribute__((ext_vector_type(4)))  int v4i_t;

#define HH   8
#define LQN  1024
#define LKVN 1024
#define DH   64
#define SPANN 128
#define NC   160          // padded key window: 5 chunks of 32
#define MROWS 2048        // B*LQ
#define KDIM 512
#define NDIM 512

__device__ __forceinline__ u16 f2bf(float f) {
    unsigned int u = __builtin_bit_cast(unsigned int, f);
    u += 0x7FFFu + ((u >> 16) & 1u);            // round-to-nearest-even
    return (u16)(u >> 16);
}

__device__ __forceinline__ void wait_tensorcnt0() {
#if __has_builtin(__builtin_amdgcn_s_wait_tensorcnt)
    __builtin_amdgcn_s_wait_tensorcnt(0);
#else
    asm volatile("s_wait_tensorcnt 0x0" ::: "memory");
#endif
}

// A-fragment: 16x32 bf16. lane m = lane%16, half = lane/16.
// lane holds K run [kk + half*16, +16)  (contiguous in row)
__device__ __forceinline__ v16bf load_a_frag(const u16* rowptr, int kk, int half) {
    const v8bf lo = *reinterpret_cast<const v8bf*>(rowptr + kk + half * 16);
    const v8bf hi = *reinterpret_cast<const v8bf*>(rowptr + kk + half * 16 + 8);
    return __builtin_shufflevector(lo, hi, 0,1,2,3,4,5,6,7,8,9,10,11,12,13,14,15);
}

// ---------------- conversion kernels ----------------
__global__ void cvt_bf16_kernel(const float* __restrict__ s, u16* __restrict__ d, int n) {
    int i = blockIdx.x * blockDim.x + threadIdx.x;
    if (i < n) d[i] = f2bf(s[i]);
}

// W (K=512 x N=512, row-major f32) -> WT (N x K, row-major bf16)
__global__ void cvtT_bf16_kernel(const float* __restrict__ W, u16* __restrict__ WT) {
    int i = blockIdx.x * blockDim.x + threadIdx.x;   // 262144 elements
    int k = i >> 9, n = i & 511;
    WT[n * KDIM + k] = f2bf(W[k * NDIM + n]);
}

// ---------------- generic 2048x512x512 WMMA GEMM ----------------
// A: 2048x512 bf16 row-major. BT: 512x512 bf16, BT[n][k] = B[k][n].
// mode 0: bf16 out at (b,h,l,64)   (Q / K projections)
// mode 1: bf16 out at (b,h,64,l)   (V projection, transposed for PV GEMM)
// mode 2: f32 out row-major 2048x512 (final output projection)
__global__ void gemm_wmma_kernel(const u16* __restrict__ A, const u16* __restrict__ BT,
                                 u16* __restrict__ outb, float* __restrict__ outf, int mode) {
    const int lane = threadIdx.x & 31;
    const int w    = threadIdx.x >> 5;
    const int lm   = lane & 15;
    const int half = lane >> 4;
    const int m0 = blockIdx.x * 64 + (w >> 1) * 16;
    const int n0 = blockIdx.y * 128 + (w & 1) * 64;

    const v8f zero = {0.f,0.f,0.f,0.f,0.f,0.f,0.f,0.f};
    v8f c[4]; for (int i = 0; i < 4; ++i) c[i] = zero;

    const u16* arow = A + (size_t)(m0 + lm) * KDIM;
    for (int kk = 0; kk < KDIM; kk += 32) {
        if (kk + 32 < KDIM) __builtin_prefetch(arow + kk + 32, 0, 0);   // global_prefetch_b8
        const v16bf a = load_a_frag(arow, kk, half);
        #pragma unroll
        for (int nt = 0; nt < 4; ++nt) {
            const int n = n0 + nt * 16 + lm;
            const v16bf b = *reinterpret_cast<const v16bf*>(BT + (size_t)n * KDIM + kk + half * 16);
            c[nt] = __builtin_amdgcn_wmma_f32_16x16x32_bf16(false, a, false, b,
                                                            (short)0, c[nt], false, false);
        }
    }

    #pragma unroll
    for (int nt = 0; nt < 4; ++nt) {
        const int n = n0 + nt * 16 + lm;
        #pragma unroll
        for (int r = 0; r < 8; ++r) {
            const int m = m0 + r + 8 * half;
            const float v = c[nt][r];
            if (mode == 2) {
                outf[(size_t)m * NDIM + n] = v;
            } else {
                const int b = m >> 10, l = m & 1023;
                const int h = n >> 6,  d = n & 63;
                const size_t idx = (mode == 0)
                    ? ((size_t)((b * HH + h) * LQN + l) * DH + d)
                    : ((size_t)((b * HH + h) * DH  + d) * LKVN + l);
                outb[idx] = f2bf(v);
            }
        }
    }
}

// ---------------- sliding-window attention ----------------
// One wave (32 threads) per 16-query tile. grid.x = B*H*(LQ/16) = 1024.
__global__ void attn_wmma_kernel(const u16* __restrict__ Qh, const u16* __restrict__ Kh,
                                 const u16* __restrict__ Vt, u16* __restrict__ outh) {
    __shared__ __align__(32) float Sx[16 * NC];      // scores
    __shared__ __align__(32) u16   Pp[16 * NC];      // probs (bf16)
    __shared__ __align__(32) u16   Vl[DH * NC];      // V window, d-major

    const int blk = blockIdx.x;
    const int qt  = blk & 63;
    const int bh  = blk >> 6;                         // b*8 + h
    const int q0  = qt * 16;
    const int kb0 = q0 - (SPANN - 1);                 // may be negative
    const int lane = threadIdx.x & 31;
    const int lm   = lane & 15;
    const int half = lane >> 4;

    // ---- stage V window into LDS as a 64x160 tile (d-major rows) ----
    const u16* vbase = Vt + (size_t)bh * DH * LKVN;
    const bool tdm_ok = (kb0 >= 0);
#if __has_builtin(__builtin_amdgcn_tensor_load_to_lds)
    if (tdm_ok) {
        // Tensor DMA descriptor (CDNA5 ISA):
        // 2D tile: tile_dim0 = NC keys (x), tile_dim1 = DH rows (y),
        // tensor_dim0 = LKVN, tensor_dim1 = DH, dim0_stride = LKVN, 2-byte elems.
        const unsigned long long gaddr = (unsigned long long)(const void*)(vbase + kb0);
        const unsigned ldsa = (unsigned)(unsigned long long)(const void*)&Vl[0];
        v4u g0 = { 1u,                                       // count=1, no gather
                   ldsa,                                     // lds_addr
                   (unsigned)(gaddr & 0xffffffffu),          // global_addr[31:0]
                   (unsigned)(((gaddr >> 32) & 0x01FFFFFFu) | (2u << 30)) }; // addr[56:32] | type=2
        v8i_t g1 = { (int)(1u << 16),                        // data_size=1 (2 bytes)
                     (int)((unsigned)LKVN << 16),            // tensor_dim0[15:0] @ bits63:48
                     (int)((unsigned)DH << 16),              // dim0 hi=0 | tensor_dim1[15:0]
                     (int)((unsigned)NC << 16),              // dim1 hi=0 | tile_dim0=160
                     (int)DH,                                // tile_dim1=64, tile_dim2=0
                     (int)LKVN,                              // tensor_dim0_stride[31:0]
                     0, 0 };
        v4i_t z4 = {0, 0, 0, 0};
        v8i_t z8 = {0, 0, 0, 0, 0, 0, 0, 0};
        __builtin_amdgcn_tensor_load_to_lds(g0, g1, z4, z4, z8, 0);
    } else
#endif
    {
        for (int idx = threadIdx.x; idx < DH * NC; idx += 32) {
            const int d = idx / NC, cc = idx % NC;
            int key = kb0 + cc;
            key = key < 0 ? 0 : (key > LKVN - 1 ? LKVN - 1 : key);
            Vl[idx] = vbase[(size_t)d * LKVN + key];
        }
    }

    // ---- scores: S(16x160) = Q(16x64) * K^T ----
    const u16* qrow  = Qh + ((size_t)bh * LQN + q0 + lm) * DH;
    const u16* kbase = Kh + (size_t)bh * LKVN * DH;
    const v8f zero = {0.f,0.f,0.f,0.f,0.f,0.f,0.f,0.f};
    v8f c[10]; for (int i = 0; i < 10; ++i) c[i] = zero;

    for (int ks = 0; ks < 2; ++ks) {
        const int kk = ks * 32;
        const v16bf a = load_a_frag(qrow, kk, half);
        #pragma unroll
        for (int nt = 0; nt < 10; ++nt) {
            int key = kb0 + nt * 16 + lm;
            key = key < 0 ? 0 : (key > LKVN - 1 ? LKVN - 1 : key);
            const v16bf b = *reinterpret_cast<const v16bf*>(kbase + (size_t)key * DH + kk + half * 16);
            c[nt] = __builtin_amdgcn_wmma_f32_16x16x32_bf16(false, a, false, b,
                                                            (short)0, c[nt], false, false);
        }
    }

    const float NEG = -3.0e38f;
    #pragma unroll
    for (int nt = 0; nt < 10; ++nt) {
        const int col = nt * 16 + lm;
        const int key = kb0 + col;
        #pragma unroll
        for (int r = 0; r < 8; ++r) {
            const int m = r + 8 * half;
            const int l = q0 + m;
            const bool valid = (key >= 0) && (key < LKVN) && (key >= l - (SPANN - 1)) && (key <= l);
            Sx[m * NC + col] = valid ? c[nt][r] * 0.125f : NEG;
        }
    }
    __syncthreads();

    // ---- softmax: 2 lanes per row, 80 cols each ----
    {
        const int row = lane >> 1;
        const int c0  = (lane & 1) * 80;
        float mx = NEG;
        for (int i = 0; i < 80; ++i) mx = fmaxf(mx, Sx[row * NC + c0 + i]);
        mx = fmaxf(mx, __shfl_xor(mx, 1, 32));
        float sum = 0.f;
        for (int i = 0; i < 80; ++i) {
            const float e = __expf(Sx[row * NC + c0 + i] - mx);
            Sx[row * NC + c0 + i] = e;
            sum += e;
        }
        sum += __shfl_xor(sum, 1, 32);
        const float rs = 1.f / sum;
        for (int i = 0; i < 80; ++i) Pp[row * NC + c0 + i] = f2bf(Sx[row * NC + c0 + i] * rs);
    }
    __syncthreads();
    wait_tensorcnt0();    // V window tile (TDM) must be resident before PV GEMM

    // ---- O(16x64) = P(16x160) * Vwin(160x64) ----
    v8f o[4]; for (int i = 0; i < 4; ++i) o[i] = zero;
    for (int ks = 0; ks < 5; ++ks) {
        const int kk = ks * 32;
        const v16bf a = load_a_frag(&Pp[lm * NC], kk, half);
        #pragma unroll
        for (int nt = 0; nt < 4; ++nt) {
            const v16bf b = *reinterpret_cast<const v16bf*>(&Vl[(nt * 16 + lm) * NC + kk + half * 16]);
            o[nt] = __builtin_amdgcn_wmma_f32_16x16x32_bf16(false, a, false, b,
                                                            (short)0, o[nt], false, false);
        }
    }

    // ---- write out_h at (b, l, h*64+d), bf16 ----
    const int b_ = bh >> 3, h_ = bh & 7;
    #pragma unroll
    for (int nt = 0; nt < 4; ++nt) {
        const int d = nt * 16 + lm;
        #pragma unroll
        for (int r = 0; r < 8; ++r) {
            const int l = q0 + r + 8 * half;
            outh[((size_t)(b_ * LQN + l)) * (HH * DH) + h_ * DH + d] = f2bf(o[nt][r]);
        }
    }
}

extern "C" void kernel_launch(void* const* d_in, const int* in_sizes, int n_in,
                              void* d_out, int out_size, void* d_ws, size_t ws_size,
                              hipStream_t stream) {
    const float* q  = (const float*)d_in[0];
    const float* k  = (const float*)d_in[1];
    const float* v  = (const float*)d_in[2];
    const float* Wq = (const float*)d_in[3];
    const float* Wk = (const float*)d_in[4];
    const float* Wv = (const float*)d_in[5];
    const float* Wo = (const float*)d_in[6];
    float* out = (float*)d_out;

    // workspace partition (bf16 buffers), 256B aligned
    char* ws = (char*)d_ws;
    size_t off = 0;
    auto alloc = [&](size_t bytes) -> u16* {
        u16* p = (u16*)(ws + off);
        off += (bytes + 255) & ~(size_t)255;
        return p;
    };
    const size_t actBytes = (size_t)MROWS * KDIM * sizeof(u16);   // 2 MB
    const size_t wBytes   = (size_t)KDIM * NDIM * sizeof(u16);    // 512 KB
    u16* qb  = alloc(actBytes);
    u16* kb  = alloc(actBytes);
    u16* vb  = alloc(actBytes);
    u16* WqT = alloc(wBytes);
    u16* WkT = alloc(wBytes);
    u16* WvT = alloc(wBytes);
    u16* WoT = alloc(wBytes);
    u16* Qh  = alloc(actBytes);   // (b,h,l,64)
    u16* Kh  = alloc(actBytes);   // (b,h,l,64)
    u16* Vt  = alloc(actBytes);   // (b,h,64,l)
    u16* Oh  = alloc(actBytes);   // (b,l,h*64+d)

    const int nAct = MROWS * KDIM;        // 1048576
    const int nW   = KDIM * NDIM;         // 262144

    cvt_bf16_kernel<<<(nAct + 255) / 256, 256, 0, stream>>>(q, qb, nAct);
    cvt_bf16_kernel<<<(nAct + 255) / 256, 256, 0, stream>>>(k, kb, nAct);
    cvt_bf16_kernel<<<(nAct + 255) / 256, 256, 0, stream>>>(v, vb, nAct);
    cvtT_bf16_kernel<<<nW / 256, 256, 0, stream>>>(Wq, WqT);
    cvtT_bf16_kernel<<<nW / 256, 256, 0, stream>>>(Wk, WkT);
    cvtT_bf16_kernel<<<nW / 256, 256, 0, stream>>>(Wv, WvT);
    cvtT_bf16_kernel<<<nW / 256, 256, 0, stream>>>(Wo, WoT);

    dim3 ggrid(MROWS / 64, NDIM / 128);
    gemm_wmma_kernel<<<ggrid, 256, 0, stream>>>(qb, WqT, Qh, nullptr, 0);
    gemm_wmma_kernel<<<ggrid, 256, 0, stream>>>(kb, WkT, Kh, nullptr, 0);
    gemm_wmma_kernel<<<ggrid, 256, 0, stream>>>(vb, WvT, Vt, nullptr, 1);

    attn_wmma_kernel<<<2 * HH * (LQN / 16), 32, 0, stream>>>(Qh, Kh, Vt, Oh);

    gemm_wmma_kernel<<<ggrid, 256, 0, stream>>>(Oh, WoT, nullptr, out, 2);
}
